// Network_45526653337928
// MI455X (gfx1250) — compile-verified, matrix operand order checked by reading, not run on hardware
//
#include <hip/hip_runtime.h>
#include <hip/hip_bf16.h>

typedef _Float16 half8  __attribute__((ext_vector_type(8)));
typedef _Float16 half16 __attribute__((ext_vector_type(16)));
typedef float    f32x8  __attribute__((ext_vector_type(8)));

#define N_RAYS_C 32768
#define N_SAMPLES_C (32 * N_RAYS_C)
#define CHUNKS_PER_BLOCK 8

// Load a 16x32-f16 WMMA operand fragment from LDS (row-major [rows][stride] halfs).
// Lane layout (ISA 7.12.2, 16-bit A-matrix 16x32): lanes 0-15 hold K={0..7,16..23},
// lanes 16-31 hold K={8..15,24..31} of row (lane&15). B uses the same layout on B^T.
__device__ __forceinline__ half16 ldsFrag(const _Float16* base, int stride, int row,
                                          int kbase, int hi8) {
  const _Float16* p = base + row * stride + kbase + hi8;
  half16 r;
  *(half8*)&r         = *(const half8*)p;         // K = kbase+hi8    .. +7
  *(((half8*)&r) + 1) = *(const half8*)(p + 16);  // K = kbase+hi8+16 .. +23
  return r;
}

__global__ __launch_bounds__(256) void nerf_mlp_kernel(
    const float* __restrict__ rays_o, const float* __restrict__ rays_d,
    const int* __restrict__ ray_indices, const float* __restrict__ t_starts,
    const float* __restrict__ t_ends,
    const float* __restrict__ base_w1, const float* __restrict__ base_b1,
    const float* __restrict__ base_w2, const float* __restrict__ base_b2,
    const float* __restrict__ head_w1, const float* __restrict__ head_b1,
    const float* __restrict__ head_w2, const float* __restrict__ head_b2,
    float* __restrict__ rgb_s, float* __restrict__ alpha_s,
    float* __restrict__ log1m_s, float* __restrict__ tmid_s)
{
  // Weights transposed to [n][k] (f16) so B fragments are contiguous-in-K loads.
  __shared__ __attribute__((aligned(16))) _Float16 sW1[64 * 64];  // base_w1: K 60->64 pad
  __shared__ __attribute__((aligned(16))) _Float16 sW2[64 * 64];  // base_w2
  __shared__ __attribute__((aligned(16))) _Float16 sW3[64 * 96];  // head_w1: K 87->96 pad
  __shared__ __attribute__((aligned(16))) _Float16 sW4[16 * 64];  // head_w2: N 3->16 pad
  __shared__ float sB1[64], sB2[64], sB3[64], sB4[16];
  __shared__ __attribute__((aligned(16))) _Float16 sAct[8][16 * 96]; // per-wave staging
  __shared__ float sDens[8][16];
  __shared__ float sRGB[8][48];

  const int tid = threadIdx.x;

  // ---- stage weights once per block (amortized over CHUNKS_PER_BLOCK chunks) ----
  for (int i = tid; i < 64 * 64; i += 256) {
    int n = i >> 6, k = i & 63;
    sW1[n * 64 + k] = (k < 60) ? (_Float16)base_w1[k * 64 + n] : (_Float16)0.f;
  }
  for (int i = tid; i < 64 * 64; i += 256) {
    int n = i >> 6, k = i & 63;
    sW2[n * 64 + k] = (_Float16)base_w2[k * 64 + n];
  }
  for (int i = tid; i < 64 * 96; i += 256) {
    int n = i / 96, k = i % 96;
    sW3[n * 96 + k] = (k < 87) ? (_Float16)head_w1[k * 64 + n] : (_Float16)0.f;
  }
  for (int i = tid; i < 16 * 64; i += 256) {
    int n = i >> 6, k = i & 63;
    sW4[n * 64 + k] = (n < 3) ? (_Float16)head_w2[k * 3 + n] : (_Float16)0.f;
  }
  if (tid < 64) { sB1[tid] = base_b1[tid]; sB2[tid] = base_b2[tid]; sB3[tid] = head_b1[tid]; }
  if (tid < 16) { sB4[tid] = (tid < 3) ? head_b2[tid] : 0.f; }
  __syncthreads();

  const int wave = tid >> 5, lane = tid & 31, lh = lane & 31 & 15;
  const bool hi = (lane >= 16);
  const int hi8 = hi ? 8 : 0;
  _Float16* act = &sAct[wave][0];
  // cos(x) = sin(x + pi/2): lanes 0-15 produce sin terms, 16-31 cos terms (no divergence)
  const float phase = hi ? 1.5707963267948966f : 0.f;
  const int encOffX = hi ? 10 : 0;  // xyz enc: sin at +0, cos at +10
  const int encOffD = hi ? 4 : 0;   // dir enc: sin at +0, cos at +4

  for (int chunk = 0; chunk < CHUNKS_PER_BLOCK; ++chunk) {
    const int samp = (blockIdx.x * CHUNKS_PER_BLOCK + chunk) * 128 + wave * 16 + lh;

    // ---- per-sample data (both half-waves load the same sample lh) ----
    int ridx = ray_indices[samp];
    float ox = rays_o[ridx * 3 + 0], oy = rays_o[ridx * 3 + 1], oz = rays_o[ridx * 3 + 2];
    float ddx = rays_d[ridx * 3 + 0], ddy = rays_d[ridx * 3 + 1], ddz = rays_d[ridx * 3 + 2];
    float ts = t_starts[samp], te = t_ends[samp];
    float tmid = 0.5f * (ts + te), dt = te - ts;

    // ---- xyz frequency encoding -> act cols 0..59, zero 60..63 ----
    {
      float p3[3] = {ox + ddx * tmid, oy + ddy * tmid, oz + ddz * tmid};
      #pragma unroll
      for (int c = 0; c < 3; c++) {
        float f = 3.14159265358979f;
        #pragma unroll
        for (int l = 0; l < 10; l++) {
          act[lh * 96 + c * 20 + encOffX + l] = (_Float16)__sinf(p3[c] * f + phase);
          f *= 2.f;
        }
      }
      if (!hi) {
        #pragma unroll
        for (int k = 60; k < 64; k++) act[lh * 96 + k] = (_Float16)0.f;
      }
    }
    // Same-wave LDS ops complete in order (DScnt): staged data is safe to read below.

    // ---- GEMM1: h = relu(enc[16x64] @ W1[64x64] + b1) ----
    {
      f32x8 acc[4]; f32x8 z = {0.f, 0.f, 0.f, 0.f, 0.f, 0.f, 0.f, 0.f};
      #pragma unroll
      for (int nt = 0; nt < 4; nt++) acc[nt] = z;
      #pragma unroll
      for (int kt = 0; kt < 2; kt++) {
        half16 a = ldsFrag(act, 96, lh, kt * 32, hi8);
        #pragma unroll
        for (int nt = 0; nt < 4; nt++) {
          half16 b = ldsFrag(sW1, 64, nt * 16 + lh, kt * 32, hi8);
          acc[nt] = __builtin_amdgcn_wmma_f32_16x16x32_f16(false, a, false, b,
                                                           (short)0, acc[nt], false, false);
        }
      }
      #pragma unroll
      for (int nt = 0; nt < 4; nt++) {
        int n = nt * 16 + lh; float bias = sB1[n];
        #pragma unroll
        for (int r = 0; r < 8; r++) {
          float v = acc[nt][r] + bias; v = v > 0.f ? v : 0.f;
          act[(r + hi8) * 96 + n] = (_Float16)v;
        }
      }
    }

    // ---- GEMM2: base_out = h @ W2 + b2 ; col0 -> density, cols1..63 -> head_in[24..86] ----
    {
      f32x8 acc[4]; f32x8 z = {0.f, 0.f, 0.f, 0.f, 0.f, 0.f, 0.f, 0.f};
      #pragma unroll
      for (int nt = 0; nt < 4; nt++) acc[nt] = z;
      #pragma unroll
      for (int kt = 0; kt < 2; kt++) {
        half16 a = ldsFrag(act, 96, lh, kt * 32, hi8);
        #pragma unroll
        for (int nt = 0; nt < 4; nt++) {
          half16 b = ldsFrag(sW2, 64, nt * 16 + lh, kt * 32, hi8);
          acc[nt] = __builtin_amdgcn_wmma_f32_16x16x32_f16(false, a, false, b,
                                                           (short)0, acc[nt], false, false);
        }
      }
      #pragma unroll
      for (int nt = 0; nt < 4; nt++) {
        int n = nt * 16 + lh; float bias = sB2[n];
        #pragma unroll
        for (int r = 0; r < 8; r++) {
          int m = r + hi8;
          float v = acc[nt][r] + bias;
          if (n == 0) sDens[wave][m] = v;
          else        act[m * 96 + 23 + n] = (_Float16)v;  // head_in cols 24..86
        }
      }
    }

    // ---- dir encoding -> head_in[0..23], pad 87..95 ----
    {
      float d3[3] = {ddx, ddy, ddz};
      #pragma unroll
      for (int c = 0; c < 3; c++) {
        float f = 3.14159265358979f;
        #pragma unroll
        for (int l = 0; l < 4; l++) {
          act[lh * 96 + c * 8 + encOffD + l] = (_Float16)__sinf(d3[c] * f + phase);
          f *= 2.f;
        }
      }
      if (!hi) {
        #pragma unroll
        for (int k = 87; k < 96; k++) act[lh * 96 + k] = (_Float16)0.f;
      }
    }

    // ---- density / alpha / log1m outputs (one lane per sample) ----
    if (!hi) {
      float raw = sDens[wave][lh];
      raw = fminf(fmaxf(raw, -15.f), 15.f);
      float dens = __expf(raw);
      float oma  = __expf(-dens * dt);
      float alpha = 1.f - oma;
      float l1m = __logf(fminf(fmaxf(oma, 1e-10f), 1.f));
      alpha_s[samp] = alpha;
      log1m_s[samp] = l1m;
      tmid_s[samp]  = tmid;
    }

    // ---- GEMM3: g = relu(head_in[16x96] @ W3[96x64] + b3) -> act cols 0..63 ----
    {
      f32x8 acc[4]; f32x8 z = {0.f, 0.f, 0.f, 0.f, 0.f, 0.f, 0.f, 0.f};
      #pragma unroll
      for (int nt = 0; nt < 4; nt++) acc[nt] = z;
      #pragma unroll
      for (int kt = 0; kt < 3; kt++) {
        half16 a = ldsFrag(act, 96, lh, kt * 32, hi8);
        #pragma unroll
        for (int nt = 0; nt < 4; nt++) {
          half16 b = ldsFrag(sW3, 96, nt * 16 + lh, kt * 32, hi8);
          acc[nt] = __builtin_amdgcn_wmma_f32_16x16x32_f16(false, a, false, b,
                                                           (short)0, acc[nt], false, false);
        }
      }
      #pragma unroll
      for (int nt = 0; nt < 4; nt++) {
        int n = nt * 16 + lh; float bias = sB3[n];
        #pragma unroll
        for (int r = 0; r < 8; r++) {
          float v = acc[nt][r] + bias; v = v > 0.f ? v : 0.f;
          act[(r + hi8) * 96 + n] = (_Float16)v;
        }
      }
    }

    // ---- GEMM4: rgb = sigmoid(g[16x64] @ W4[64x16pad] + b4), cols 0..2 valid ----
    {
      f32x8 acc = {0.f, 0.f, 0.f, 0.f, 0.f, 0.f, 0.f, 0.f};
      #pragma unroll
      for (int kt = 0; kt < 2; kt++) {
        half16 a = ldsFrag(act, 96, lh, kt * 32, hi8);
        half16 b = ldsFrag(sW4, 64, lh, kt * 32, hi8);
        acc = __builtin_amdgcn_wmma_f32_16x16x32_f16(false, a, false, b,
                                                     (short)0, acc, false, false);
      }
      if (lh < 3) {
        #pragma unroll
        for (int r = 0; r < 8; r++) {
          int m = r + hi8;
          float v = acc[r] + sB4[lh];
          sRGB[wave][m * 3 + lh] = 1.f / (1.f + __expf(-v));
        }
      }
    }
    if (!hi) {
      rgb_s[samp * 3 + 0] = sRGB[wave][lh * 3 + 0];
      rgb_s[samp * 3 + 1] = sRGB[wave][lh * 3 + 1];
      rgb_s[samp * 3 + 2] = sRGB[wave][lh * 3 + 2];
    }
  }
}

// ---------------- compositing: scan + segment reductions ----------------

__global__ __launch_bounds__(256) void zero_out_kernel(float* __restrict__ out) {
  int i = blockIdx.x * 256 + threadIdx.x;
  if (i < 5 * N_RAYS_C) out[i] = 0.f;
}

__global__ __launch_bounds__(256) void scan_block_kernel(const float* __restrict__ in,
                                                         float* __restrict__ incl,
                                                         float* __restrict__ bsums) {
  __shared__ float sh[256];
  int t = threadIdx.x;
  int base = blockIdx.x * 1024 + t * 4;
  float a0 = in[base + 0], a1 = in[base + 1], a2 = in[base + 2], a3 = in[base + 3];
  float s1 = a0 + a1, s2 = s1 + a2, s3 = s2 + a3;
  sh[t] = s3;
  __syncthreads();
  for (int o = 1; o < 256; o <<= 1) {
    float y = (t >= o) ? sh[t - o] : 0.f;
    __syncthreads();
    sh[t] += y;
    __syncthreads();
  }
  float excl = sh[t] - s3;
  incl[base + 0] = excl + a0; incl[base + 1] = excl + s1;
  incl[base + 2] = excl + s2; incl[base + 3] = excl + s3;
  if (t == 255) bsums[blockIdx.x] = sh[255];
}

__global__ __launch_bounds__(1024) void scan_sums_kernel(float* __restrict__ bsums) {
  __shared__ float sh[1024];
  int t = threadIdx.x;
  float v = bsums[t];
  sh[t] = v;
  __syncthreads();
  for (int o = 1; o < 1024; o <<= 1) {
    float y = (t >= o) ? sh[t - o] : 0.f;
    __syncthreads();
    sh[t] += y;
    __syncthreads();
  }
  bsums[t] = sh[t] - v;  // exclusive block offset
}

__global__ __launch_bounds__(256) void cexcl_segfirst_kernel(
    const float* __restrict__ log1m_s, float* __restrict__ incl,
    const float* __restrict__ bsums, const int* __restrict__ ray_indices,
    float* __restrict__ segfirst) {
  int i = blockIdx.x * 256 + threadIdx.x;
  float ce = bsums[i >> 10] + incl[i] - log1m_s[i];
  incl[i] = ce;  // now holds c_excl
  int r = ray_indices[i];
  // sorted indices + non-increasing c_excl => segment_max == first element of segment
  if (i == 0 || ray_indices[i - 1] != r) segfirst[r] = ce;
}

__global__ __launch_bounds__(256) void composite_kernel(
    const float* __restrict__ cexcl, const float* __restrict__ segfirst,
    const float* __restrict__ alpha_s, const float* __restrict__ rgb_s,
    const float* __restrict__ tmid_s, const int* __restrict__ ray_indices,
    float* __restrict__ out) {
  int i = blockIdx.x * 256 + threadIdx.x;
  int r = ray_indices[i];
  float w = __expf(cexcl[i] - segfirst[r]) * alpha_s[i];
  out[5 * N_RAYS_C + i] = w;  // weights output
  atomicAdd(&out[r * 3 + 0], w * rgb_s[i * 3 + 0]);
  atomicAdd(&out[r * 3 + 1], w * rgb_s[i * 3 + 1]);
  atomicAdd(&out[r * 3 + 2], w * rgb_s[i * 3 + 2]);
  atomicAdd(&out[3 * N_RAYS_C + r], w);              // opacity
  atomicAdd(&out[4 * N_RAYS_C + r], w * tmid_s[i]);  // depth
}

extern "C" void kernel_launch(void* const* d_in, const int* in_sizes, int n_in,
                              void* d_out, int out_size, void* d_ws, size_t ws_size,
                              hipStream_t stream) {
  const float* rays_o      = (const float*)d_in[0];
  const float* rays_d      = (const float*)d_in[1];
  const int*   ray_indices = (const int*)  d_in[2];
  const float* t_starts    = (const float*)d_in[3];
  const float* t_ends      = (const float*)d_in[4];
  const float* base_w1     = (const float*)d_in[5];
  const float* base_b1     = (const float*)d_in[6];
  const float* base_w2     = (const float*)d_in[7];
  const float* base_b2     = (const float*)d_in[8];
  const float* head_w1     = (const float*)d_in[9];
  const float* head_b1     = (const float*)d_in[10];
  const float* head_w2     = (const float*)d_in[11];
  const float* head_b2     = (const float*)d_in[12];

  const int N = N_SAMPLES_C;
  float* wsf      = (float*)d_ws;
  float* rgb_s    = wsf;            // 3N
  float* alpha_s  = wsf + 3 * (size_t)N;
  float* log1m_s  = wsf + 4 * (size_t)N;
  float* tmid_s   = wsf + 5 * (size_t)N;
  float* incl     = wsf + 6 * (size_t)N;
  float* bsums    = wsf + 7 * (size_t)N;        // 1024
  float* segfirst = wsf + 7 * (size_t)N + 1024; // N_RAYS
  float* out = (float*)d_out;

  zero_out_kernel<<<(5 * N_RAYS_C + 255) / 256, 256, 0, stream>>>(out);
  nerf_mlp_kernel<<<N / (128 * CHUNKS_PER_BLOCK), 256, 0, stream>>>(
      rays_o, rays_d, ray_indices, t_starts, t_ends,
      base_w1, base_b1, base_w2, base_b2,
      head_w1, head_b1, head_w2, head_b2,
      rgb_s, alpha_s, log1m_s, tmid_s);
  scan_block_kernel<<<N / 1024, 256, 0, stream>>>(log1m_s, incl, bsums);
  scan_sums_kernel<<<1, 1024, 0, stream>>>(bsums);
  cexcl_segfirst_kernel<<<N / 256, 256, 0, stream>>>(log1m_s, incl, bsums, ray_indices, segfirst);
  composite_kernel<<<N / 256, 256, 0, stream>>>(incl, segfirst, alpha_s, rgb_s, tmid_s, ray_indices, out);
  (void)in_sizes; (void)n_in; (void)out_size; (void)ws_size;
}
